// DecoderCell_77988016161121
// MI455X (gfx1250) — matrix operand order repaired and checked by reading, not compiled, not run on hardware
//
#include <hip/hip_runtime.h>
#include <hip/hip_bf16.h>
#include <math.h>

// ---------------------------------------------------------------------------
// Sizes
// ---------------------------------------------------------------------------
#define B    128
#define FR   1024
#define DEC  512
#define ATT  512
#define LIS  256
#define VOC  123
#define RNNI 1147          // DEC + VOC + 2*LIS
#define RNNI_PAD 1152      // padded to multiple of 32
#define CCH  8             // context-reduction frame chunks

typedef __attribute__((ext_vector_type(16))) _Float16 v16h;
typedef __attribute__((ext_vector_type(8)))  float    v8f;
typedef __attribute__((ext_vector_type(4)))  unsigned u32x4;

// ---------------------------------------------------------------------------
// Small hash RNG (stateless, deterministic)
// ---------------------------------------------------------------------------
__device__ __forceinline__ unsigned wang_hash(unsigned x) {
    x = (x ^ 61u) ^ (x >> 16);
    x *= 9u;
    x ^= x >> 4;
    x *= 0x27d4eb2du;
    x ^= x >> 15;
    return x;
}
__device__ __forceinline__ unsigned hash3(unsigned a, unsigned b, unsigned c) {
    return wang_hash(a * 0x9E3779B1u ^ wang_hash(b * 0x85EBCA77u ^ wang_hash(c)));
}
__device__ __forceinline__ float u01(unsigned h) {
    return (float)(h >> 8) * (1.0f / 16777216.0f);
}

// ---------------------------------------------------------------------------
// Scheduled sampling: gumbel-argmax over pyp logits, blend with yin
// ---------------------------------------------------------------------------
__global__ __launch_bounds__(128) void sched_sample_kernel(
    const float* __restrict__ yin, const float* __restrict__ pyp,
    const unsigned char* __restrict__ blend, float* __restrict__ ytu)
{
    __shared__ float sval[128];
    __shared__ int   sidx[128];
    const int b = blockIdx.x, tid = threadIdx.x;

    float best = -INFINITY; int bi = 0;
    for (int v = tid; v < VOC; v += 128) {
        unsigned h = hash3(42u, (unsigned)b, (unsigned)v);
        float u = fmaxf(u01(h), 1e-12f);
        float g = -__logf(-__logf(u));
        float s = pyp[b * VOC + v] + g;
        if (s > best) { best = s; bi = v; }
    }
    sval[tid] = best; sidx[tid] = bi;
    __syncthreads();
    for (int s = 64; s > 0; s >>= 1) {
        if (tid < s && sval[tid + s] > sval[tid]) {
            sval[tid] = sval[tid + s]; sidx[tid] = sidx[tid + s];
        }
        __syncthreads();
    }
    const int arg = sidx[0];
    const float rut = u01(hash3(1337u, (unsigned)b, 7u));
    const float bl  = blend[0] ? 1.0f : 0.0f;
    for (int v = tid; v < VOC; v += 128) {
        float yv = yin[b * VOC + v];
        float smp = (v == arg) ? 1.0f : 0.0f;
        float yt  = (rut < 0.1f) ? smp : yv;
        ytu[b * VOC + v] = bl * yt + (1.0f - bl) * yv;
    }
}

// ---------------------------------------------------------------------------
// Pack up to 3 f32 sources (concat along K) into an f16 [B, Kpad] matrix.
// ---------------------------------------------------------------------------
__global__ __launch_bounds__(256) void pack3_kernel(
    _Float16* __restrict__ dst, int Kpad,
    const float* __restrict__ s0, int l0,
    const float* __restrict__ s1, int l1,
    const float* __restrict__ s2, int l2)
{
    int idx = blockIdx.x * 256 + threadIdx.x;
    int total = B * Kpad;
    if (idx >= total) return;
    int r = idx / Kpad, c = idx - r * Kpad;
    float v = 0.0f;
    if (c < l0)                 v = s0[r * l0 + c];
    else if (c < l0 + l1)       v = s1[r * l1 + (c - l0)];
    else if (c < l0 + l1 + l2)  v = s2[r * l2 + (c - l0 - l1)];
    dst[idx] = (_Float16)v;
}

// ---------------------------------------------------------------------------
// WMMA GEMM:  C[128,N] = A1[128,K1]·W1[K1act,N] + A2[128,K2]·W2[K2act,N]+bias
// A* pre-packed f16 (row stride = Kpad); W* f32 in global, converted to f16
// while staging to LDS in a per-lane packed-pair layout.  A tile is copied
// global->LDS with async b128 loads (ASYNCcnt).  8 waves/block, one 16x16
// WMMA tile each; all operand gathers are ds_load_b128.
// ---------------------------------------------------------------------------
__global__ __launch_bounds__(256) void gemm_wmma_kernel(
    const _Float16* __restrict__ A1, int K1pad, const float* __restrict__ W1, int K1act,
    const _Float16* __restrict__ A2, int K2pad, const float* __restrict__ W2, int K2act,
    const float* __restrict__ bias, float* __restrict__ C, int N, int relu)
{
    __shared__ alignas(16) _Float16 sA[128 * 32]; // A tile: 128 rows x 32 k (8KB)
    __shared__ alignas(16) unsigned sBp[256];     // B tile packed: [hi][n][v] k-pairs

    const int tid  = threadIdx.x;
    const int lane = tid & 31;
    const int wave = tid >> 5;          // 0..7 -> M rows [16w, 16w+16)
    const int m    = lane & 15;
    const int hi   = lane >> 4;
    const int n0   = blockIdx.x * 16;

    const unsigned* sAu = (const unsigned*)sA;
    const unsigned ldsA = (unsigned)(unsigned long long)(const void*)sA;
    const int rowbase = (wave * 16 + m) * 16;     // uint index of this lane's A row

    // B staging decode for this thread (one packed word per thread)
    const int sb_hi = tid >> 7;
    const int sb_n  = (tid >> 3) & 15;
    const int sb_v  = tid & 7;
    const int sb_ng = n0 + sb_n;

    v8f acc = {};

#pragma unroll 1
    for (int seg = 0; seg < 2; ++seg) {
        const _Float16* A  = seg ? A2 : A1;
        const float*    Wg = seg ? W2 : W1;
        const int Kpad = seg ? K2pad : K1pad;
        const int Kact = seg ? K2act : K1act;
        if (A == nullptr || Kpad == 0) continue;

        for (int k0 = 0; k0 < Kpad; k0 += 32) {
            // ---- A tile: async global->LDS, 512 x 16B chunks ----
            {
                const char* Ab = (const char*)A + (size_t)k0 * 2;
                for (int c = tid; c < 512; c += 256) {
                    unsigned lo = ldsA + (unsigned)c * 16u;
                    const void* ga = Ab + (size_t)(c >> 2) * ((size_t)Kpad * 2)
                                        + (size_t)(c & 3) * 16;
                    asm volatile("global_load_async_to_lds_b128 %0, %1, off"
                                 :: "v"(lo), "v"(ga) : "memory");
                }
            }
            // ---- B tile: f32 -> f16 packed pairs [hi][n][v] ----
            {
                int kg0 = k0 + sb_hi * 16 + sb_v * 2;
                union { _Float16 h[2]; unsigned u; } p;
                p.h[0] = (_Float16)((kg0     < Kact && sb_ng < N)
                                    ? Wg[(size_t)kg0 * N + sb_ng] : 0.0f);
                p.h[1] = (_Float16)((kg0 + 1 < Kact && sb_ng < N)
                                    ? Wg[(size_t)(kg0 + 1) * N + sb_ng] : 0.0f);
                sBp[tid] = p.u;
            }
            asm volatile("s_wait_asynccnt 0x0" ::: "memory");
            __syncthreads();

            // ---- operand fragments: 4x ds_load_b128 per lane ----
            union { v16h v; u32x4 q[2]; } af, bf;
            af.q[0] = *(const u32x4*)&sAu[rowbase + hi * 8];
            af.q[1] = *(const u32x4*)&sAu[rowbase + hi * 8 + 4];
            bf.q[0] = *(const u32x4*)&sBp[(hi * 16 + m) * 8];
            bf.q[1] = *(const u32x4*)&sBp[(hi * 16 + m) * 8 + 4];

            acc = __builtin_amdgcn_wmma_f32_16x16x32_f16(
                false, af.v, false, bf.v, (short)0, acc, false, false);
            __syncthreads();
        }
    }

    // C/D layout: lanes 0-15 vgpr r -> M=r ; lanes 16-31 vgpr r -> M=8+r
    const int col = n0 + m;
    if (col < N) {
        float bia = bias ? bias[col] : 0.0f;
#pragma unroll
        for (int r = 0; r < 8; ++r) {
            int row = wave * 16 + hi * 8 + r;
            float v = acc[r] + bia;
            if (relu) v = fmaxf(v, 0.0f);
            C[(size_t)row * N + col] = v;
        }
    }
}

// ---------------------------------------------------------------------------
// LSTM elementwise (Keras gate order i,f,g,o)
// ---------------------------------------------------------------------------
__global__ __launch_bounds__(256) void lstm_ew_kernel(
    const float* __restrict__ z, const float* __restrict__ c_prev,
    float* __restrict__ h_out, float* __restrict__ c_out,
    float* __restrict__ h_out2, _Float16* __restrict__ h_h16)
{
    int idx = blockIdx.x * 256 + threadIdx.x;
    if (idx >= B * DEC) return;
    int b = idx >> 9, j = idx & (DEC - 1);
    const float* zr = z + (size_t)b * 4 * DEC;
    float zi = zr[j], zf = zr[j + DEC], zg = zr[j + 2 * DEC], zo = zr[j + 3 * DEC];
    float gi = 1.0f / (1.0f + __expf(-zi));
    float gf = 1.0f / (1.0f + __expf(-zf));
    float gg = tanhf(zg);
    float go = 1.0f / (1.0f + __expf(-zo));
    float cn = gf * c_prev[idx] + gi * gg;
    float hn = go * tanhf(cn);
    h_out[idx] = hn;
    c_out[idx] = cn;
    if (h_out2) h_out2[idx] = hn;
    h_h16[idx] = (_Float16)hn;
}

// ---------------------------------------------------------------------------
// Attention scores: one wave per (b, frame); 512-dim dot with b128 loads.
// ---------------------------------------------------------------------------
__global__ __launch_bounds__(256) void att_scores_kernel(
    const float* __restrict__ q, const float* __restrict__ keys,
    const unsigned char* __restrict__ mask, const float* __restrict__ scale,
    float* __restrict__ scores)
{
    int gwave = blockIdx.x * 8 + (threadIdx.x >> 5);
    int lane  = threadIdx.x & 31;
    int b = gwave >> 10;            // FR = 1024
    int f = gwave & (FR - 1);
    const float4* kr = (const float4*)(keys + ((size_t)b * FR + f) * ATT);
    const float4* qr = (const float4*)(q + (size_t)b * ATT);
    float s = 0.0f;
#pragma unroll
    for (int i = 0; i < 4; ++i) {
        float4 kv = kr[lane + i * 32];
        float4 qv = qr[lane + i * 32];
        s += kv.x * qv.x + kv.y * qv.y + kv.z * qv.z + kv.w * qv.w;
    }
#pragma unroll
    for (int off = 16; off > 0; off >>= 1) s += __shfl_down(s, off, 32);
    if (lane == 0) {
        s *= scale[0];
        if (!mask[b * FR + f]) s = -1e9f;
        scores[(size_t)b * FR + f] = s;
    }
}

// ---------------------------------------------------------------------------
// Softmax over FR per row (in place -> weights), mean att index, loss terms.
// ---------------------------------------------------------------------------
__global__ __launch_bounds__(256) void att_softmax_kernel(
    float* __restrict__ scores, const float* __restrict__ yin,
    const float* __restrict__ pmai, const float* __restrict__ ploss,
    float* __restrict__ nmai_out, float* __restrict__ nloss_out)
{
    __shared__ float red[256];
    const int b = blockIdx.x, tid = threadIdx.x;
    float* row = scores + (size_t)b * FR;

    float mx = -INFINITY;
#pragma unroll
    for (int i = 0; i < 4; ++i) mx = fmaxf(mx, row[tid + i * 256]);
    red[tid] = mx; __syncthreads();
    for (int s = 128; s > 0; s >>= 1) {
        if (tid < s) red[tid] = fmaxf(red[tid], red[tid + s]);
        __syncthreads();
    }
    mx = red[0]; __syncthreads();

    float e[4], sum = 0.0f;
#pragma unroll
    for (int i = 0; i < 4; ++i) { e[i] = __expf(row[tid + i * 256] - mx); sum += e[i]; }
    red[tid] = sum; __syncthreads();
    for (int s = 128; s > 0; s >>= 1) {
        if (tid < s) red[tid] += red[tid + s];
        __syncthreads();
    }
    sum = red[0]; __syncthreads();

    const float inv = 1.0f / sum;
    float wsum = 0.0f;
#pragma unroll
    for (int i = 0; i < 4; ++i) {
        int f = tid + i * 256;
        float a = e[i] * inv;
        row[f] = a;
        wsum += a * (float)f;
    }
    red[tid] = wsum; __syncthreads();
    for (int s = 128; s > 0; s >>= 1) {
        if (tid < s) red[tid] += red[tid + s];
        __syncthreads();
    }
    if (tid == 0) {
        float nmai = red[0];
        float pm = pmai[b];
        float diff = fminf(fabsf(nmai - pm), 1.0f);
        float losst = diff * ((nmai < pm) ? 1.0f : 0.0f)
                    - diff * ((nmai > pm) ? 1.0f : 0.0f);
        float ym = 0.0f;
        for (int v = 0; v < VOC; ++v) ym += yin[b * VOC + v];
        nmai_out[b]  = nmai;
        nloss_out[b] = ym * losst + ploss[b];
    }
}

// ---------------------------------------------------------------------------
// Context, stage 1: partial ci over a 128-frame chunk; float4 streaming.
// grid = (B, CCH), 128 threads (4 channels each).
// ---------------------------------------------------------------------------
__global__ __launch_bounds__(128) void att_context_partial_kernel(
    const float* __restrict__ saw, const float* __restrict__ feats,
    float* __restrict__ partial)  // [B][CCH][512]
{
    const int b = blockIdx.x, ch = blockIdx.y, d4 = threadIdx.x;
    const int FPC = FR / CCH;   // 128 frames per chunk
    const float4* fb = (const float4*)(feats + ((size_t)b * FR + (size_t)ch * FPC) * (2 * LIS));
    const float* sw = saw + (size_t)b * FR + (size_t)ch * FPC;
    float4 acc = make_float4(0.f, 0.f, 0.f, 0.f);
#pragma unroll 4
    for (int f = 0; f < FPC; ++f) {
        float w = sw[f];
        float4 v = fb[(size_t)f * 128 + d4];
        acc.x += w * v.x; acc.y += w * v.y; acc.z += w * v.z; acc.w += w * v.w;
    }
    ((float4*)partial)[((size_t)b * CCH + ch) * 128 + d4] = acc;
}

// Context, stage 2: reduce chunks -> ci[b,d]
__global__ __launch_bounds__(256) void att_context_reduce_kernel(
    const float* __restrict__ partial, float* __restrict__ ci)
{
    int idx = blockIdx.x * 256 + threadIdx.x;     // B * 512
    if (idx >= B * 2 * LIS) return;
    int b = idx >> 9, d = idx & 511;
    float s = 0.0f;
#pragma unroll
    for (int c = 0; c < CCH; ++c)
        s += partial[((size_t)b * CCH + c) * (2 * LIS) + d];
    ci[idx] = s;
}

// ---------------------------------------------------------------------------
// Host launcher
// ---------------------------------------------------------------------------
extern "C" void kernel_launch(void* const* d_in, const int* in_sizes, int n_in,
                              void* d_out, int out_size, void* d_ws, size_t ws_size,
                              hipStream_t stream) {
    (void)in_sizes; (void)n_in; (void)out_size; (void)ws_size;

    const float* yin   = (const float*)d_in[0];
    const float* h1    = (const float*)d_in[1];
    const float* c1    = (const float*)d_in[2];
    const float* h2    = (const float*)d_in[3];
    const float* c2    = (const float*)d_in[4];
    const float* psv   = (const float*)d_in[5];
    const float* pcv   = (const float*)d_in[6];
    const float* pmai  = (const float*)d_in[7];
    const float* ploss = (const float*)d_in[8];
    const float* pyp   = (const float*)d_in[9];
    const float* lfeat = (const float*)d_in[10];
    const float* lkeys = (const float*)d_in[11];
    const unsigned char* lmask = (const unsigned char*)d_in[12];
    const unsigned char* blend = (const unsigned char*)d_in[13];
    const float* W1 = (const float*)d_in[14];
    const float* U1 = (const float*)d_in[15];
    const float* b1 = (const float*)d_in[16];
    const float* W2 = (const float*)d_in[17];
    const float* U2 = (const float*)d_in[18];
    const float* b2 = (const float*)d_in[19];
    const float* phi1_w = (const float*)d_in[20];
    const float* phi1_b = (const float*)d_in[21];
    const float* phi2_w = (const float*)d_in[22];
    const float* phi2_b = (const float*)d_in[23];
    const float* chr1_w = (const float*)d_in[24];
    const float* chr1_b = (const float*)d_in[25];
    const float* chr2_w = (const float*)d_in[26];
    const float* chr2_b = (const float*)d_in[27];
    const float* att_scale = (const float*)d_in[28];

    float* out = (float*)d_out;
    float* o_yp    = out;                      // [B,VOC]
    float* o_h1n   = out + 15744;              // [B,DEC]
    float* o_c1n   = out + 81280;
    float* o_h2n   = out + 146816;
    float* o_c2n   = out + 212352;
    float* o_si    = out + 277888;
    float* o_ci    = out + 343424;
    float* o_nmai  = out + 408960;             // [B]
    float* o_nloss = out + 409088;             // [B]

    char* ws = (char*)d_ws;
    size_t off = 0;
    auto alloc = [&](size_t bytes) -> char* {
        char* p = ws + off;
        off = (off + bytes + 255) & ~(size_t)255;
        return p;
    };
    float*     w_ytu    = (float*)alloc((size_t)B * VOC * 4);
    _Float16*  w_rnni_h = (_Float16*)alloc((size_t)B * RNNI_PAD * 2);
    _Float16*  w_h1_h   = (_Float16*)alloc((size_t)B * DEC * 2);
    _Float16*  w_h2_h   = (_Float16*)alloc((size_t)B * DEC * 2);
    float*     w_z      = (float*)alloc((size_t)B * 4 * DEC * 4);
    _Float16*  w_h1n_h  = (_Float16*)alloc((size_t)B * DEC * 2);
    _Float16*  w_si_h   = (_Float16*)alloc((size_t)B * DEC * 2);
    float*     w_m1     = (float*)alloc((size_t)B * 1024 * 4);
    _Float16*  w_m1_h   = (_Float16*)alloc((size_t)B * 1024 * 2);
    float*     w_m2     = (float*)alloc((size_t)B * ATT * 4);
    float*     w_scores = (float*)alloc((size_t)B * FR * 4);
    float*     w_part   = (float*)alloc((size_t)B * CCH * 2 * LIS * 4);
    _Float16*  w_sc_h   = (_Float16*)alloc((size_t)B * 1024 * 2);
    float*     w_ch     = (float*)alloc((size_t)B * 492 * 4);
    _Float16*  w_ch_h   = (_Float16*)alloc((size_t)B * 512 * 2);

    const int PACK_T = 256;
    auto packBlocks = [](int kpad) { return (B * kpad + 255) / 256; };

    // 1) scheduled sampling -> ytu
    sched_sample_kernel<<<B, 128, 0, stream>>>(yin, pyp, blend, w_ytu);

    // 2) pack A operands
    pack3_kernel<<<packBlocks(RNNI_PAD), PACK_T, 0, stream>>>(
        w_rnni_h, RNNI_PAD, psv, DEC, w_ytu, VOC, pcv, 2 * LIS);
    pack3_kernel<<<packBlocks(DEC), PACK_T, 0, stream>>>(
        w_h1_h, DEC, h1, DEC, nullptr, 0, nullptr, 0);
    pack3_kernel<<<packBlocks(DEC), PACK_T, 0, stream>>>(
        w_h2_h, DEC, h2, DEC, nullptr, 0, nullptr, 0);

    // 3) LSTM1
    gemm_wmma_kernel<<<(4 * DEC) / 16, 256, 0, stream>>>(
        w_rnni_h, RNNI_PAD, W1, RNNI, w_h1_h, DEC, U1, DEC, b1, w_z, 4 * DEC, 0);
    lstm_ew_kernel<<<(B * DEC) / 256, 256, 0, stream>>>(
        w_z, c1, o_h1n, o_c1n, nullptr, w_h1n_h);

    // 4) LSTM2 (h2n == si)
    gemm_wmma_kernel<<<(4 * DEC) / 16, 256, 0, stream>>>(
        w_h1n_h, DEC, W2, DEC, w_h2_h, DEC, U2, DEC, b2, w_z, 4 * DEC, 0);
    lstm_ew_kernel<<<(B * DEC) / 256, 256, 0, stream>>>(
        w_z, c2, o_h2n, o_c2n, o_si, w_si_h);

    // 5) query MLP
    gemm_wmma_kernel<<<1024 / 16, 256, 0, stream>>>(
        w_si_h, DEC, phi1_w, DEC, nullptr, 0, nullptr, 0, phi1_b, w_m1, 1024, 1);
    pack3_kernel<<<packBlocks(1024), PACK_T, 0, stream>>>(
        w_m1_h, 1024, w_m1, 1024, nullptr, 0, nullptr, 0);
    gemm_wmma_kernel<<<ATT / 16, 256, 0, stream>>>(
        w_m1_h, 1024, phi2_w, 1024, nullptr, 0, nullptr, 0, phi2_b, w_m2, ATT, 0);

    // 6) attention
    att_scores_kernel<<<(B * FR) / 8, 256, 0, stream>>>(
        w_m2, lkeys, lmask, att_scale, w_scores);
    att_softmax_kernel<<<B, 256, 0, stream>>>(
        w_scores, yin, pmai, ploss, o_nmai, o_nloss);
    att_context_partial_kernel<<<dim3(B, CCH), 128, 0, stream>>>(
        w_scores, lfeat, w_part);
    att_context_reduce_kernel<<<(B * 2 * LIS) / 256, 256, 0, stream>>>(w_part, o_ci);

    // 7) char MLP
    pack3_kernel<<<packBlocks(1024), PACK_T, 0, stream>>>(
        w_sc_h, 1024, o_si, DEC, o_ci, 2 * LIS, nullptr, 0);
    gemm_wmma_kernel<<<(492 + 15) / 16, 256, 0, stream>>>(
        w_sc_h, 1024, chr1_w, 1024, nullptr, 0, nullptr, 0, chr1_b, w_ch, 492, 1);
    pack3_kernel<<<packBlocks(512), PACK_T, 0, stream>>>(
        w_ch_h, 512, w_ch, 492, nullptr, 0, nullptr, 0);
    gemm_wmma_kernel<<<(VOC + 15) / 16, 256, 0, stream>>>(
        w_ch_h, 512, chr2_w, 492, nullptr, 0, nullptr, 0, chr2_b, o_yp, VOC, 0);
}
